// EncoderLayer_3805341024813
// MI455X (gfx1250) — matrix operand order repaired and checked by reading, not compiled, not run on hardware
//
#include <hip/hip_runtime.h>
#include <hip/hip_bf16.h>
#include <math.h>

// ---------------------------------------------------------------------------
// Transformer encoder layer for MI455X (gfx1250), bf16 WMMA matrix cores.
// B=4, S=2048, D=512, H=8, HD=64.  All matmuls run on v_wmma_f32_16x16x32_bf16
// with f32 accumulation; LN / softmax / epilogues in f32 VALU.
// ---------------------------------------------------------------------------

typedef __bf16 bf16;
typedef __attribute__((ext_vector_type(16))) __bf16 v16bf;
typedef __attribute__((ext_vector_type(8)))  __bf16 v8bf;
typedef __attribute__((ext_vector_type(8)))  float  v8f;

#define DD   512
#define SS   2048
#define BB   4
#define HH   8
#define HD   64
#define NROWS (BB * SS)   // 8192

// ---- WMMA wrapper ----------------------------------------------------------
__device__ __forceinline__ v8f wmma_bf16(v16bf a, v16bf b, v8f c) {
  // (neg_a, A, neg_b, B, c_mod, C, reuse_a, reuse_b)
  return __builtin_amdgcn_wmma_f32_16x16x32_bf16(false, a, false, b, (short)0, c,
                                                 false, false);
}

// ---- Fragment loaders (layouts per CDNA5 ISA §7.12.2) -----------
// A operand 16x32 bf16: lane m = lane&15 is the row; halfwave g = lane>>4;
// elements 0..7 -> k = k0 + 8g + e ; elements 8..15 -> k = k0 + 16 + 8g + (e-8)
__device__ __forceinline__ v16bf load_a_frag(const bf16* __restrict__ p,
                                             int row0, int ld, int k0) {
  const int lane = threadIdx.x & 31;
  const int m = lane & 15, g = lane >> 4;
  const bf16* a = p + (size_t)(row0 + m) * ld + (k0 + 8 * g);
  v8bf lo = *reinterpret_cast<const v8bf*>(a);
  v8bf hi = *reinterpret_cast<const v8bf*>(a + 16);
  v16bf f;
#pragma unroll
  for (int i = 0; i < 8; ++i) { f[i] = lo[i]; f[i + 8] = hi[i]; }
  return f;
}

// B operand 32x16 bf16: lane n = lane&15 is the column; halfwave g = lane>>4;
// elements 0..7 -> k = k0 + 8g + e ; elements 8..15 -> k = k0 + 16 + 8g + (e-8).
// Source stored so that the k-dimension is contiguous per column (i.e. B[k][n] = src[n][k], row-major).
__device__ __forceinline__ v16bf load_b_frag(const bf16* __restrict__ p,
                                             int col0, int ld, int k0) {
  const int lane = threadIdx.x & 31;
  const int n = lane & 15, g = lane >> 4;
  const bf16* a = p + (size_t)(col0 + n) * ld + (k0 + 8 * g);
  v8bf lo = *reinterpret_cast<const v8bf*>(a);
  v8bf hi = *reinterpret_cast<const v8bf*>(a + 16);
  v16bf f;
#pragma unroll
  for (int i = 0; i < 8; ++i) { f[i] = lo[i]; f[i + 8] = hi[i]; }
  return f;
}
// C/D 16x16 f32: element v, lane l -> row = v + 8*(l>>4), col = l&15.

// ---- weight f32 -> bf16 ----------------------------------------------------
__global__ __launch_bounds__(256) void cvt_kernel(const float* __restrict__ in,
                                                  bf16* __restrict__ out, int n) {
  int i = blockIdx.x * 256 + threadIdx.x;
  if (i < n) out[i] = (bf16)in[i];
}

// ---- LayerNorm: one wave per row, torch semantics: a*(x-mean)/(std+eps)+b,
//      std unbiased (ddof=1).  Optionally stores f32 copy for residual. ------
template <bool STORE_F32>
__global__ __launch_bounds__(256) void ln_kernel(const float* __restrict__ x,
                                                 const float* __restrict__ ga,
                                                 const float* __restrict__ be,
                                                 float* __restrict__ outf,
                                                 bf16* __restrict__ outb) {
  const int lane = threadIdx.x & 31;
  const int row = blockIdx.x * 8 + (threadIdx.x >> 5);
  const float* xr = x + (size_t)row * DD;
  float vals[16];
  float s = 0.f, ss = 0.f;
#pragma unroll
  for (int i = 0; i < 16; ++i) {
    float v = xr[lane + 32 * i];
    vals[i] = v; s += v; ss += v * v;
  }
#pragma unroll
  for (int m = 16; m >= 1; m >>= 1) {
    s  += __shfl_xor(s,  m, 32);
    ss += __shfl_xor(ss, m, 32);
  }
  const float mean = s * (1.f / DD);
  const float var  = (ss - (float)DD * mean * mean) * (1.f / (DD - 1));
  const float inv  = 1.f / (sqrtf(var) + 1e-6f);
#pragma unroll
  for (int i = 0; i < 16; ++i) {
    int c = lane + 32 * i;
    float y = ga[c] * (vals[i] - mean) * inv + be[c];
    if (STORE_F32) outf[(size_t)row * DD + c] = y;
    outb[(size_t)row * DD + c] = (bf16)y;
  }
}

// ---- Generic GEMM: C[M,512] = A[M,512](bf16) @ W[512,512](bf16, torch
//      [out,in] layout == ready-made B operand).  Wave: 16 rows x 64 cols. ---
enum { EPI_BF16 = 0, EPI_VT = 1, EPI_ADDRES_F32 = 2,
       EPI_RELU_BIAS_BF16 = 3, EPI_BIAS_ADD_F32OUT = 4 };

template <int EPI>
__global__ __launch_bounds__(256) void gemm_kernel(const bf16* __restrict__ A,
                                                   const bf16* __restrict__ W,
                                                   const float* __restrict__ bias,
                                                   const float* __restrict__ res,
                                                   bf16* __restrict__ outb,
                                                   float* __restrict__ outf) {
  const int lane = threadIdx.x & 31;
  const int wave = threadIdx.x >> 5;
  const int row0 = blockIdx.x * 128 + wave * 16;
  const int col0 = blockIdx.y * 64;
  v8f acc[4] = {};
#pragma unroll 2
  for (int k0 = 0; k0 < DD; k0 += 32) {
    v16bf a = load_a_frag(A, row0, DD, k0);
#pragma unroll
    for (int t = 0; t < 4; ++t) {
      v16bf b = load_b_frag(W, col0 + t * 16, DD, k0);
      acc[t] = wmma_bf16(a, b, acc[t]);
    }
  }
  const int g = lane >> 4, n = lane & 15;
#pragma unroll
  for (int t = 0; t < 4; ++t) {
#pragma unroll
    for (int v = 0; v < 8; ++v) {
      const int r = row0 + v + 8 * g;
      const int c = col0 + t * 16 + n;
      float x = acc[t][v];
      if (EPI == EPI_BF16) {
        outb[(size_t)r * DD + c] = (bf16)x;
      } else if (EPI == EPI_VT) {           // V stored head-transposed [B][H][HD][S]
        const int b_ = r >> 11, s_ = r & (SS - 1);
        const int h_ = c >> 6,  hd_ = c & (HD - 1);
        outb[(((size_t)(b_ * HH + h_)) * HD + hd_) * SS + s_] = (bf16)x;
      } else if (EPI == EPI_ADDRES_F32) {   // h = xn + attn@wo^T
        outf[(size_t)r * DD + c] = x + res[(size_t)r * DD + c];
      } else if (EPI == EPI_RELU_BIAS_BF16) {
        float y = x + bias[c];
        outb[(size_t)r * DD + c] = (bf16)(y > 0.f ? y : 0.f);
      } else {                              // out = h + ffn@w2^T + b2
        outf[(size_t)r * DD + c] = x + bias[c] + res[(size_t)r * DD + c];
      }
    }
  }
}

// ---- Flash attention: one wave per (b, h, 16-query tile). ------------------
// Computes S^T = K·Q^T so the score C-layout converts per-lane (no shuffles)
// into the A-fragment of P for the P·V WMMA.
__global__ __launch_bounds__(256) void attn_kernel(const bf16* __restrict__ Q,
                                                   const bf16* __restrict__ K,
                                                   const bf16* __restrict__ Vt,
                                                   bf16* __restrict__ O) {
  const int lane  = threadIdx.x & 31;
  const int wg    = blockIdx.x * 8 + (threadIdx.x >> 5);  // 0..4095
  const int qtile = wg & 127;
  const int h     = (wg >> 7) & 7;
  const int b     = wg >> 10;
  const int q0    = qtile * 16;

  const bf16* qp = Q  + (size_t)b * SS * DD + h * HD;
  const bf16* kp = K  + (size_t)b * SS * DD + h * HD;
  const bf16* vp = Vt + ((size_t)(b * HH + h)) * HD * SS;

  // Q as B operand (per-lane column = query, k = head dim, contiguous)
  const v16bf bq0 = load_b_frag(qp, q0, DD, 0);
  const v16bf bq1 = load_b_frag(qp, q0, DD, 32);

  v8f o[4] = {};
  float mrun = -1e30f, lrun = 0.f;
  const int g8 = (lane >> 4) * 8;

  for (int t0 = 0; t0 < SS; t0 += 32) {
    // S^T tiles: rows = keys (VGPR axis), cols = queries (lane axis)
    v8f s0 = {}, s1 = {};
    s0 = wmma_bf16(load_a_frag(kp, t0,      DD, 0 ), bq0, s0);
    s0 = wmma_bf16(load_a_frag(kp, t0,      DD, 32), bq1, s0);
    s1 = wmma_bf16(load_a_frag(kp, t0 + 16, DD, 0 ), bq0, s1);
    s1 = wmma_bf16(load_a_frag(kp, t0 + 16, DD, 32), bq1, s1);

    float mc = -1e30f;
#pragma unroll
    for (int v = 0; v < 8; ++v) {
      s0[v] *= 0.125f;  // 1/sqrt(64)
      s1[v] *= 0.125f;
      mc = fmaxf(mc, fmaxf(s0[v], s1[v]));
    }
    mc = fmaxf(mc, __shfl_xor(mc, 16, 32));   // combine key halves per query
    const float mnew  = fmaxf(mrun, mc);
    const float alpha = __expf(mrun - mnew);

    // exp() -> P, laid out directly as the next WMMA's A fragment
    v16bf pa;
    float lc = 0.f;
#pragma unroll
    for (int v = 0; v < 8; ++v) {
      const float p0 = __expf(s0[v] - mnew);   // key k = v + 8g      (rel.)
      const float p1 = __expf(s1[v] - mnew);   // key k = 16 + v + 8g (rel.)
      lc += p0 + p1;
      pa[v]     = (bf16)p0;
      pa[v + 8] = (bf16)p1;
    }
    lc += __shfl_xor(lc, 16, 32);
    lrun = lrun * alpha + lc;
    mrun = mnew;

    // rescale O accumulators: row (query) lives on the VGPR axis -> broadcast
#pragma unroll
    for (int v = 0; v < 8; ++v) {
      const float ar = __shfl(alpha, v + g8, 32);
#pragma unroll
      for (int t = 0; t < 4; ++t) o[t][v] *= ar;
    }

    // O += P @ V  (V head-transposed: per-lane column = hd, k = key, contig.)
#pragma unroll
    for (int t = 0; t < 4; ++t)
      o[t] = wmma_bf16(pa, load_b_frag(vp, t * 16, SS, t0), o[t]);
  }

  // normalize by row sums and store attn output [b][s][d] as bf16
  const int n = lane & 15;
#pragma unroll
  for (int v = 0; v < 8; ++v) {
    const float linv = 1.f / __shfl(lrun, v + g8, 32);
    const int srow = q0 + v + g8;
#pragma unroll
    for (int t = 0; t < 4; ++t)
      O[((size_t)b * SS + srow) * DD + h * HD + t * 16 + n] =
          (bf16)(o[t][v] * linv);
  }
}

// ---------------------------------------------------------------------------
extern "C" void kernel_launch(void* const* d_in, const int* in_sizes, int n_in,
                              void* d_out, int out_size, void* d_ws, size_t ws_size,
                              hipStream_t stream) {
  const float* x     = (const float*)d_in[0];
  const float* ln1_a = (const float*)d_in[1];
  const float* ln1_b = (const float*)d_in[2];
  const float* ln2_a = (const float*)d_in[3];
  const float* ln2_b = (const float*)d_in[4];
  const float* wq = (const float*)d_in[5];
  const float* wk = (const float*)d_in[6];
  const float* wv = (const float*)d_in[7];
  const float* wo = (const float*)d_in[8];
  const float* w1 = (const float*)d_in[9];
  const float* b1 = (const float*)d_in[10];
  const float* w2 = (const float*)d_in[11];
  const float* b2 = (const float*)d_in[12];
  float* out = (float*)d_out;

  // workspace carve-up
  char* ws = (char*)d_ws;
  size_t off = 0;
  auto take = [&](size_t bytes) -> void* {
    void* p = ws + off;
    off += (bytes + 255) & ~(size_t)255;
    return p;
  };
  const size_t WBYTES = (size_t)DD * DD * sizeof(bf16);
  const size_t ABYTES = (size_t)NROWS * DD * sizeof(bf16);
  const size_t FBYTES = (size_t)NROWS * DD * sizeof(float);
  bf16* wq_b = (bf16*)take(WBYTES);
  bf16* wk_b = (bf16*)take(WBYTES);
  bf16* wv_b = (bf16*)take(WBYTES);
  bf16* wo_b = (bf16*)take(WBYTES);
  bf16* w1_b = (bf16*)take(WBYTES);
  bf16* w2_b = (bf16*)take(WBYTES);
  float* xn_f = (float*)take(FBYTES);
  float* h_f  = (float*)take(FBYTES);
  bf16* xn_b  = (bf16*)take(ABYTES);
  bf16* q_b   = (bf16*)take(ABYTES);
  bf16* k_b   = (bf16*)take(ABYTES);
  bf16* vt_b  = (bf16*)take(ABYTES);
  bf16* at_b  = (bf16*)take(ABYTES);
  bf16* hn_b  = (bf16*)take(ABYTES);
  bf16* f1_b  = (bf16*)take(ABYTES);

  const int WN = DD * DD;                       // 262144
  const dim3 cvtG((WN + 255) / 256), blk(256);
  cvt_kernel<<<cvtG, blk, 0, stream>>>(wq, wq_b, WN);
  cvt_kernel<<<cvtG, blk, 0, stream>>>(wk, wk_b, WN);
  cvt_kernel<<<cvtG, blk, 0, stream>>>(wv, wv_b, WN);
  cvt_kernel<<<cvtG, blk, 0, stream>>>(wo, wo_b, WN);
  cvt_kernel<<<cvtG, blk, 0, stream>>>(w1, w1_b, WN);
  cvt_kernel<<<cvtG, blk, 0, stream>>>(w2, w2_b, WN);

  const dim3 lnG(NROWS / 8);                    // 8 rows (waves) per block
  ln_kernel<true><<<lnG, blk, 0, stream>>>(x, ln1_a, ln1_b, xn_f, xn_b);

  const dim3 gemmG(NROWS / 128, DD / 64);       // 64 x 8 blocks
  gemm_kernel<EPI_BF16><<<gemmG, blk, 0, stream>>>(xn_b, wq_b, nullptr, nullptr, q_b, nullptr);
  gemm_kernel<EPI_BF16><<<gemmG, blk, 0, stream>>>(xn_b, wk_b, nullptr, nullptr, k_b, nullptr);
  gemm_kernel<EPI_VT  ><<<gemmG, blk, 0, stream>>>(xn_b, wv_b, nullptr, nullptr, vt_b, nullptr);

  const dim3 attnG(BB * HH * (SS / 16) / 8);    // 4096 waves / 8 per block
  attn_kernel<<<attnG, blk, 0, stream>>>(q_b, k_b, vt_b, at_b);

  gemm_kernel<EPI_ADDRES_F32><<<gemmG, blk, 0, stream>>>(at_b, wo_b, nullptr, xn_f, nullptr, h_f);

  ln_kernel<false><<<lnG, blk, 0, stream>>>(h_f, ln2_a, ln2_b, nullptr, hn_b);

  gemm_kernel<EPI_RELU_BIAS_BF16><<<gemmG, blk, 0, stream>>>(hn_b, w1_b, b1, nullptr, f1_b, nullptr);
  gemm_kernel<EPI_BIAS_ADD_F32OUT><<<gemmG, blk, 0, stream>>>(f1_b, w2_b, b2, h_f, nullptr, out);
}
